// SPAN_B_89335319757574
// MI455X (gfx1250) — matrix operand order repaired and checked by reading, not compiled, not run on hardware
//
#include <hip/hip_runtime.h>
#include <hip/hip_bf16.h>

typedef _Float16 h16;
typedef __attribute__((ext_vector_type(16))) _Float16 v16h;
typedef __attribute__((ext_vector_type(8)))  _Float16 v8h;
typedef __attribute__((ext_vector_type(8)))  float    v8f;

#define GB    16
#define GN    1024
#define GDIN  128
#define GDOUT 32

// ---------------- concat(x,l) -> f16 ----------------
__global__ __launch_bounds__(256)
void k_concat(const float* __restrict__ x, const float* __restrict__ l,
              h16* __restrict__ out, long rows)
{
  long idx = (long)blockIdx.x * 256 + threadIdx.x;
  long total = rows * GDIN;
  if (idx >= total) return;
  long r = idx >> 7;            // /128
  int  c = (int)(idx & 127);
  float v = (c < 64) ? x[r * 64 + c] : l[r * 64 + (c - 64)];
  out[idx] = (h16)v;
}

// ---------------- generic WMMA GEMM ----------------
// C[M,N] = act(A[M,K] @ B + bias), A is f16 row-major.
// BT=false: B stored [K,N] row-major. BT=true: B stored [N,K] (torch Linear weight).
// ACT: 0=none, 1=relu, 2=elu. Batched via grid.z with element strides.
// MT/NTW: 16x16 WMMA tiles per wave along M/N. Block = 8 waves as 4(M) x 2(N).
// Block tile = (64*MT) x (32*NTW). Double-buffered LDS, 1 barrier per K-step.
template<typename TB, typename TC, bool BT, int ACT, bool BIAS, int MT, int NTW>
__global__ __launch_bounds__(256)
void k_gemm_wmma(const h16* __restrict__ A, const TB* __restrict__ Bm,
                 const float* __restrict__ bias, TC* __restrict__ C,
                 int M, int N, int K, long sA, long sB, long sC)
{
  constexpr int BM = 64 * MT;
  constexpr int BN = 32 * NTW;
  __shared__ __align__(32) h16 As[2][BM][32];   // [m][k]
  __shared__ __align__(32) h16 Bs[2][BN][32];   // [n][k]
  const int bz = blockIdx.z;
  A  += (long)bz * sA;
  Bm += (long)bz * sB;
  C  += (long)bz * sC;
  const int n0   = blockIdx.x * BN;
  const int m0   = blockIdx.y * BM;
  const int t    = threadIdx.x;
  const int lane = t & 31;
  const int wave = t >> 5;        // 8 waves
  const int wm   = wave & 3;      // 4 along M
  const int wn   = wave >> 2;     // 2 along N

  v8f acc[MT][NTW];
  #pragma unroll
  for (int mi = 0; mi < MT; ++mi)
    #pragma unroll
    for (int ni = 0; ni < NTW; ++ni)
      acc[mi][ni] = (v8f){0.f,0.f,0.f,0.f,0.f,0.f,0.f,0.f};

  auto stage = [&](int buf, int k0) {
    { // A tile BMx32: contiguous chunks per thread, clamped rows (uniform flow)
      constexpr int AC  = (BM * 32) / 256;   // halves per thread
      constexpr int CPR = 32 / AC;           // chunks per row
      int row = t / CPR;
      int col = (t % CPR) * AC;
      int gm = m0 + row; if (gm >= M) gm = M - 1;
      const h16* src = A + (long)gm * K + k0 + col;
      #pragma unroll
      for (int i = 0; i < AC; ++i) As[buf][row][col + i] = src[i];
    }
    if (BT) { // B stored [N,K]: k-contiguous reads, contiguous LDS writes
      constexpr int BC   = (BN * 32) / 256;
      constexpr int BCPR = 32 / BC;
      int nn = t / BCPR;
      int kk = (t % BCPR) * BC;
      const TB* src = Bm + (long)(n0 + nn) * K + k0 + kk;
      #pragma unroll
      for (int i = 0; i < BC; ++i) Bs[buf][nn][kk + i] = (h16)(float)src[i];
    } else {  // B stored [K,N]: n-contiguous (coalesced) reads, transpose into LDS
      constexpr int NC = BN / 8;             // n's per thread
      int kk = t >> 3;                       // 32 k-rows, 8 threads each
      int ng = (t & 7) * NC;
      const TB* src = Bm + (long)(k0 + kk) * N + n0 + ng;
      #pragma unroll
      for (int i = 0; i < NC; ++i) Bs[buf][ng + i][kk] = (h16)(float)src[i];
    }
  };

  stage(0, 0);
  __syncthreads();
  const int nsteps = K >> 5;
  for (int it = 0; it < nsteps; ++it) {
    const int cur = it & 1;
    if (it + 1 < nsteps) stage(cur ^ 1, (it + 1) << 5);   // overlap with compute

    // fragment loads: lane half g holds k = 16g..16g+15 (contiguous) for both A and B
    const int g  = lane >> 4;
    const int mr = lane & 15;
    v16h af[MT], bf[NTW];
    #pragma unroll
    for (int mi = 0; mi < MT; ++mi) {
      af[mi] = *(const v16h*)&As[cur][wm * 16 * MT + mi * 16 + mr][g * 16];
    }
    #pragma unroll
    for (int ni = 0; ni < NTW; ++ni)
      bf[ni] = *(const v16h*)&Bs[cur][wn * 16 * NTW + ni * 16 + mr][g * 16];

    #pragma unroll
    for (int mi = 0; mi < MT; ++mi)
      #pragma unroll
      for (int ni = 0; ni < NTW; ++ni)
        acc[mi][ni] = __builtin_amdgcn_wmma_f32_16x16x32_f16(
            false, af[mi], false, bf[ni], (short)0, acc[mi][ni], false, false);
    __syncthreads();
  }

  // epilogue: C/D layout — lane col = lane&15; VGPR r -> row r + 8*(lane>>4)
  const int nr = lane & 15;
  const int rg = (lane >> 4) * 8;
  #pragma unroll
  for (int mi = 0; mi < MT; ++mi) {
    const int rbase = m0 + wm * 16 * MT + mi * 16 + rg;
    #pragma unroll
    for (int ni = 0; ni < NTW; ++ni) {
      const int n = n0 + wn * 16 * NTW + ni * 16 + nr;
      float bv = BIAS ? bias[n] : 0.f;
      #pragma unroll
      for (int r = 0; r < 8; ++r) {
        int m = rbase + r;
        if (m < M) {
          float v = acc[mi][ni][r] + bv;
          if (ACT == 1)      v = fmaxf(v, 0.f);
          else if (ACT == 2) v = (v > 0.f) ? v : (__expf(v) - 1.f);
          C[(long)m * N + n] = (TC)v;
        }
      }
    }
  }
}

// ---------------- e_src / e_dst row dots ----------------
__global__ __launch_bounds__(256)
void k_rowdot2(const h16* __restrict__ Wh, const float* __restrict__ a,
               float* __restrict__ esrc, float* __restrict__ edst,
               long rows, int K)
{
  long w = ((long)blockIdx.x * blockDim.x + threadIdx.x) >> 5;
  int lane = threadIdx.x & 31;
  if (w >= rows) return;
  const h16* p = Wh + w * K;
  float s0 = 0.f, s1 = 0.f;
  for (int k = lane; k < K; k += 32) {
    float v = (float)p[k];
    s0 += v * a[k];
    s1 += v * a[K + k];
  }
  #pragma unroll
  for (int off = 16; off > 0; off >>= 1) {
    s0 += __shfl_xor(s0, off, 32);
    s1 += __shfl_xor(s1, off, 32);
  }
  if (lane == 0) { esrc[w] = s0; edst[w] = s1; }
}

// ---------------- masked leaky-relu softmax row ----------------
__global__ __launch_bounds__(256)
void k_attn(const float* __restrict__ esrc, const float* __restrict__ edst,
            const int* __restrict__ adj, h16* __restrict__ att)
{
  const long row = blockIdx.x;          // b*GN + i
  const long b   = row >> 10;           // GN == 1024
  const int  t   = threadIdx.x;
  const int* arow = adj + row * GN;
  const float ei  = esrc[row];
  const float* ed = edst + b * GN;
  __shared__ float red[256];
  float vals[4];
  float lmax = -3.0e38f;
  #pragma unroll
  for (int jj = 0; jj < 4; ++jj) {
    int j = t + jj * 256;
    float e = ei + ed[j];
    e = (e > 0.f) ? e : 0.2f * e;        // leaky_relu alpha=0.2
    e = (arow[j] > 0) ? e : -9.0e15f;    // mask
    vals[jj] = e;
    lmax = fmaxf(lmax, e);
  }
  red[t] = lmax; __syncthreads();
  for (int s = 128; s > 0; s >>= 1) { if (t < s) red[t] = fmaxf(red[t], red[t + s]); __syncthreads(); }
  float mx = red[0]; __syncthreads();
  float lsum = 0.f;
  #pragma unroll
  for (int jj = 0; jj < 4; ++jj) { vals[jj] = __expf(vals[jj] - mx); lsum += vals[jj]; }
  red[t] = lsum; __syncthreads();
  for (int s = 128; s > 0; s >>= 1) { if (t < s) red[t] += red[t + s]; __syncthreads(); }
  float inv = 1.f / red[0];
  h16* orow = att + row * GN;
  #pragma unroll
  for (int jj = 0; jj < 4; ++jj) orow[t + jj * 256] = (h16)(vals[jj] * inv);
}

extern "C" void kernel_launch(void* const* d_in, const int* in_sizes, int n_in,
                              void* d_out, int out_size, void* d_ws, size_t ws_size,
                              hipStream_t stream)
{
  const float* x    = (const float*)d_in[0];
  const float* l    = (const float*)d_in[1];
  const int*   adj  = (const int*)d_in[2];
  const float* fc_w = (const float*)d_in[3];
  const float* fc_b = (const float*)d_in[4];
  const float* W1   = (const float*)d_in[5];
  const float* a1   = (const float*)d_in[6];
  const float* W2   = (const float*)d_in[7];
  const float* a2   = (const float*)d_in[8];
  const float* fc1w = (const float*)d_in[9];
  const float* fc1b = (const float*)d_in[10];
  const float* fc2w = (const float*)d_in[11];
  const float* fc2b = (const float*)d_in[12];
  float* out = (float*)d_out;

  char* ws = (char*)d_ws;
  size_t off = 0;
  auto take = [&](size_t bytes) { char* p = ws + off; off += (bytes + 255) & ~(size_t)255; return p; };
  const long ROWS = (long)GB * GN;   // 16384

  h16*  hcat = (h16*)take(ROWS * GDIN  * sizeof(h16));
  h16*  h0   = (h16*)take(ROWS * GDIN  * sizeof(h16));
  h16*  Wh1  = (h16*)take(ROWS * GN    * sizeof(h16));
  h16*  att  = (h16*)take(ROWS * GN    * sizeof(h16));   // reused for att1 and att2
  h16*  h1   = (h16*)take(ROWS * GN    * sizeof(h16));
  h16*  Wh2  = (h16*)take(ROWS * GDOUT * sizeof(h16));
  h16*  h2   = (h16*)take(ROWS * GDOUT * sizeof(h16));
  h16*  z    = (h16*)take((long)GB * GN * sizeof(h16));
  float* es1 = (float*)take(ROWS * sizeof(float));
  float* ed1 = (float*)take(ROWS * sizeof(float));
  float* es2 = (float*)take(ROWS * sizeof(float));
  float* ed2 = (float*)take(ROWS * sizeof(float));

  dim3 blk(256);

  // h = concat(x,l) (f16)
  {
    long total = ROWS * GDIN;
    k_concat<<<dim3((unsigned)((total + 255) / 256)), blk, 0, stream>>>(x, l, hcat, ROWS);
  }
  // h0 = relu(h @ fc_w^T + fc_b)          [16384,128]   BIG tile 128x64
  k_gemm_wmma<float, h16, true, 1, true, 2, 2><<<dim3(GDIN/64, ROWS/128, 1), blk, 0, stream>>>(
      hcat, fc_w, fc_b, h0, (int)ROWS, GDIN, GDIN, 0, 0, 0);
  // Wh1 = h0 @ W1                          [16384,1024]  BIG
  k_gemm_wmma<float, h16, false, 0, false, 2, 2><<<dim3(GN/64, ROWS/128, 1), blk, 0, stream>>>(
      h0, W1, nullptr, Wh1, (int)ROWS, GN, GDIN, 0, 0, 0);
  // e_src1/e_dst1
  k_rowdot2<<<dim3(ROWS/8), blk, 0, stream>>>(Wh1, a1, es1, ed1, ROWS, GN);
  // att1 = softmax(mask(leaky_relu(esrc+edst^T)))
  k_attn<<<dim3((unsigned)ROWS), blk, 0, stream>>>(es1, ed1, adj, att);
  // h1 = elu(att1 @ Wh1)   batched 16x [1024,1024]@[1024,1024]  (dominant GEMM)  BIG
  k_gemm_wmma<h16, h16, false, 2, false, 2, 2><<<dim3(GN/64, GN/128, GB), blk, 0, stream>>>(
      att, Wh1, nullptr, h1, GN, GN, GN, (long)GN*GN, (long)GN*GN, (long)GN*GN);
  // Wh2 = h1 @ W2                          [16384,32]    SMALL tile 64x32
  k_gemm_wmma<float, h16, false, 0, false, 1, 1><<<dim3(GDOUT/32, ROWS/64, 1), blk, 0, stream>>>(
      h1, W2, nullptr, Wh2, (int)ROWS, GDOUT, GN, 0, 0, 0);
  // e_src2/e_dst2
  k_rowdot2<<<dim3(ROWS/8), blk, 0, stream>>>(Wh2, a2, es2, ed2, ROWS, GDOUT);
  // att2
  k_attn<<<dim3((unsigned)ROWS), blk, 0, stream>>>(es2, ed2, adj, att);
  // h2 = elu(att2 @ Wh2)   batched 16x [1024,1024]@[1024,32]   SMALL
  k_gemm_wmma<h16, h16, false, 2, false, 1, 1><<<dim3(GDOUT/32, GN/64, GB), blk, 0, stream>>>(
      att, Wh2, nullptr, h2, GN, GDOUT, GN, (long)GN*GN, (long)GN*GDOUT, (long)GN*GDOUT);
  // z = relu(h2.reshape(16,32768) @ fc1_w^T + fc1_b)   [16,1024]  BIG (M clamped)
  k_gemm_wmma<float, h16, true, 1, true, 2, 2><<<dim3(GN/64, 1, 1), blk, 0, stream>>>(
      h2, fc1w, fc1b, z, GB, GN, GN*GDOUT, 0, 0, 0);
  // y = z @ fc2_w^T + fc2_b                [16,32] -> d_out (fp32)  SMALL
  k_gemm_wmma<float, float, true, 0, true, 1, 1><<<dim3(GDOUT/32, 1, 1), blk, 0, stream>>>(
      z, fc2w, fc2b, out, GB, GDOUT, GN, 0, 0, 0);

  (void)in_sizes; (void)n_in; (void)out_size; (void)ws_size;
}
